// Attention_layer_89936615178679
// MI455X (gfx1250) — matrix operand order repaired and checked by reading, not compiled, not run on hardware
//
#include <hip/hip_runtime.h>
#include <hip/hip_bf16.h>

// ---------------------------------------------------------------------------
// GAT layer for MI455X (gfx1250, wave32, WMMA).
//
// Sizes (compile-time constants from the reference):
#define N_NODES 50000
#define IN_FEAT 256
#define HEADS   8
#define OUT_F   64
#define COLS    (HEADS * OUT_F)   // 512
//
// Workspace layout (bytes, all offsets 256B aligned):
//   Wb   : bf16 copy of Ws,  HEADS*OUT_F*IN_FEAT ushorts   = 262,144 B
//   t    : f32  projections, HEADS*N_NODES*OUT_F floats    = 102,400,000 B (L2-resident)
//   Al   : f32  HEADS*N_NODES                              = 1,600,000 B
//   Ar   : f32  HEADS*N_NODES
//   m    : f32  HEADS*N_NODES  (segment max)
//   den  : f32  HEADS*N_NODES  (segment sum of exp)
// ---------------------------------------------------------------------------

typedef __attribute__((ext_vector_type(16))) __bf16 v16bf;
typedef __attribute__((ext_vector_type(8)))  float  v8f;

// float -> bf16 bits with round-to-nearest-even
__device__ __forceinline__ unsigned short f2bf_bits(float f) {
    unsigned u = __float_as_uint(f);
    u += 0x7FFFu + ((u >> 16) & 1u);
    return (unsigned short)(u >> 16);
}

// float atomic max via signed/unsigned int atomics (hardware atomics, no CAS loop)
__device__ __forceinline__ void atomicMaxF(float* addr, float val) {
    if (val >= 0.0f) atomicMax((int*)addr, __float_as_int(val));
    else             atomicMin((unsigned int*)addr, (unsigned int)__float_as_int(val));
}

// ---------------------------------------------------------------------------
// Kernel 0: convert Ws (f32) -> bf16 workspace copy
__global__ void gat_convW(const float* __restrict__ Ws, unsigned short* __restrict__ Wb) {
    int tid = blockIdx.x * blockDim.x + threadIdx.x;
    if (tid < HEADS * OUT_F * IN_FEAT) Wb[tid] = f2bf_bits(Ws[tid]);
}

// Kernel 1: zero output accumulator, init m=-inf, den=0
__global__ void gat_init(float* __restrict__ out, float* __restrict__ m, float* __restrict__ den) {
    int tid = blockIdx.x * blockDim.x + threadIdx.x;
    if (tid < N_NODES * COLS) out[tid] = 0.0f;
    if (tid < HEADS * N_NODES) { m[tid] = -INFINITY; den[tid] = 0.0f; }
}

// ---------------------------------------------------------------------------
// Kernel 2: t[h,n,o] = sum_f x[n,f] * Ws[h,o,f]  via v_wmma_f32_16x16x32_bf16.
// Block = 256 threads = 8 waves. Block owns 16 node-rows x all 512 columns.
// Wave w owns column tiles {w, w+8, w+16, w+24} (16 cols each), so the A
// fragment is reused across 4 WMMAs per K-chunk.
__global__ __launch_bounds__(256) void gat_gemm(const float* __restrict__ x,
                                                const unsigned short* __restrict__ Wb,
                                                float* __restrict__ t) {
    __shared__ unsigned short lx[16 * IN_FEAT];   // 8 KB bf16 x-tile

    const int tid   = threadIdx.x;
    const int node0 = blockIdx.x * 16;            // 3125 * 16 == 50000 exactly

    // Stage + convert the 16x256 x tile (coalesced f32 loads, bf16 LDS stores)
    #pragma unroll
    for (int ii = 0; ii < 16; ++ii) {
        int idx = ii * 256 + tid;                 // 4096 elements
        int row = idx >> 8, col = idx & 255;
        lx[idx] = f2bf_bits(x[(size_t)(node0 + row) * IN_FEAT + col]);
    }
    __syncthreads();

    const int wave = tid >> 5;
    const int lane = tid & 31;
    const int half = lane >> 4;    // ISA: lanes 0-15 / 16-31 split
    const int mr   = lane & 15;    // A: row M; B/C/D: column N

    v8f acc[4] = {v8f{}, v8f{}, v8f{}, v8f{}};

    for (int kc = 0; kc < IN_FEAT; kc += 32) {
        // A fragment (16x32 bf16): per lane 16 contiguous elems:
        //   K = kc + 16*half + {0..15}
        v16bf a;
        const unsigned short* ap = &lx[mr * IN_FEAT + kc + 16 * half];
        __builtin_memcpy(&a, ap, 16);
        __builtin_memcpy((char*)&a + 16, ap + 8, 16);

        #pragma unroll
        for (int cc = 0; cc < 4; ++cc) {
            // B fragment (32x16 bf16): column c = head*64+o, per lane 16
            // contiguous bf16 along F starting at kc + 16*half -> one 32B load.
            int c = (wave + 8 * cc) * 16 + mr;
            v16bf b;
            __builtin_memcpy(&b, &Wb[(size_t)c * IN_FEAT + kc + 16 * half], 32);
            acc[cc] = __builtin_amdgcn_wmma_f32_16x16x32_bf16(
                false, a, false, b, (short)0, acc[cc], false, false);
        }
    }

    // D layout: v8f element r holds row (r + 8*half), column mr
    #pragma unroll
    for (int cc = 0; cc < 4; ++cc) {
        int c = (wave + 8 * cc) * 16 + mr;
        int h = c >> 6, o = c & 63;
        #pragma unroll
        for (int r = 0; r < 8; ++r) {
            int node = node0 + r + 8 * half;
            t[((size_t)h * N_NODES + node) * OUT_F + o] = acc[cc][r];
        }
    }
}

// ---------------------------------------------------------------------------
// Kernel 3: Al[h,n] = t[h,n,:].a_l[h];  Ar[h,n] = t[h,n,:].a_r[h]
__global__ void gat_alar(const float* __restrict__ t, const float* __restrict__ As,
                         float* __restrict__ Al, float* __restrict__ Ar) {
    int tid = blockIdx.x * blockDim.x + threadIdx.x;
    if (tid >= HEADS * N_NODES) return;
    int h = tid / N_NODES;
    const float* tr = t + (size_t)tid * OUT_F;    // tid == h*N + n
    const float* al = As + h * 2 * OUT_F;
    const float* ar = al + OUT_F;
    float sl = 0.f, sr = 0.f;
    #pragma unroll 8
    for (int o = 0; o < OUT_F; ++o) { float v = tr[o]; sl += v * al[o]; sr += v * ar[o]; }
    Al[tid] = sl; Ar[tid] = sr;
}

// Kernel 4: segment max of e = Al[src] + Ar[dst] over src
__global__ void gat_emax(const int* __restrict__ src, const int* __restrict__ dst,
                         const float* __restrict__ Al, const float* __restrict__ Ar,
                         float* __restrict__ m, int E) {
    int tid = blockIdx.x * blockDim.x + threadIdx.x;
    if (tid >= HEADS * E) return;
    int h = tid / E, i = tid - h * E;
    int s = src[i], d = dst[i];
    float e = Al[h * N_NODES + s] + Ar[h * N_NODES + d];
    atomicMaxF(&m[h * N_NODES + s], e);
}

// Kernel 5: den[h,src] += exp(e - m[h,src])
__global__ void gat_eden(const int* __restrict__ src, const int* __restrict__ dst,
                         const float* __restrict__ Al, const float* __restrict__ Ar,
                         const float* __restrict__ m, float* __restrict__ den, int E) {
    int tid = blockIdx.x * blockDim.x + threadIdx.x;
    if (tid >= HEADS * E) return;
    int h = tid / E, i = tid - h * E;
    int s = src[i], d = dst[i];
    int hs = h * N_NODES + s;
    float e = Al[hs] + Ar[h * N_NODES + d];
    unsafeAtomicAdd(&den[hs], __expf(e - m[hs]));
}

// Kernel 6: out[src, h*64+o] += alpha * t[h, dst, o]
// One block per edge; wave = head; each lane handles 2 output columns.
__global__ __launch_bounds__(256) void gat_aggr(const int* __restrict__ src,
                                                const int* __restrict__ dst,
                                                const float* __restrict__ Al,
                                                const float* __restrict__ Ar,
                                                const float* __restrict__ m,
                                                const float* __restrict__ den,
                                                const float* __restrict__ t,
                                                float* __restrict__ out, int E) {
    int i = blockIdx.x;
    if (i >= E) return;
    int h    = threadIdx.x >> 5;
    int lane = threadIdx.x & 31;
    int s = src[i], d = dst[i];
    int hs = h * N_NODES + s;
    float e     = Al[hs] + Ar[h * N_NODES + d];
    float alpha = __expf(e - m[hs]) / den[hs];
    const float2 tv = *(const float2*)(t + ((size_t)h * N_NODES + d) * OUT_F + lane * 2);
    float* op = out + (size_t)s * COLS + h * OUT_F + lane * 2;
    unsafeAtomicAdd(op,     alpha * tv.x);
    unsafeAtomicAdd(op + 1, alpha * tv.y);
}

// Kernel 7: in-place ELU
__global__ void gat_elu(float* __restrict__ out) {
    int tid = blockIdx.x * blockDim.x + threadIdx.x;
    if (tid >= N_NODES * COLS) return;
    float v = out[tid];
    out[tid] = v > 0.0f ? v : (__expf(v) - 1.0f);
}

// ---------------------------------------------------------------------------
extern "C" void kernel_launch(void* const* d_in, const int* in_sizes, int n_in,
                              void* d_out, int out_size, void* d_ws, size_t ws_size,
                              hipStream_t stream) {
    const float* x   = (const float*)d_in[0];
    const int*   src = (const int*)d_in[1];
    const int*   dst = (const int*)d_in[2];
    const float* Ws  = (const float*)d_in[3];
    const float* As  = (const float*)d_in[4];
    float* out = (float*)d_out;
    const int E = in_sizes[1];

    // Workspace carve-up
    char* ws = (char*)d_ws;
    size_t off = 0;
    unsigned short* Wb = (unsigned short*)(ws + off); off += (size_t)HEADS * OUT_F * IN_FEAT * 2;
    off = (off + 255) & ~(size_t)255;
    float* t   = (float*)(ws + off); off += (size_t)HEADS * N_NODES * OUT_F * 4;
    float* Al  = (float*)(ws + off); off += (size_t)HEADS * N_NODES * 4;
    float* Ar  = (float*)(ws + off); off += (size_t)HEADS * N_NODES * 4;
    float* m   = (float*)(ws + off); off += (size_t)HEADS * N_NODES * 4;
    float* den = (float*)(ws + off); off += (size_t)HEADS * N_NODES * 4;

    const int BT = 256;
    gat_convW<<<(HEADS * OUT_F * IN_FEAT + BT - 1) / BT, BT, 0, stream>>>(Ws, Wb);
    gat_init <<<(N_NODES * COLS + BT - 1) / BT, BT, 0, stream>>>(out, m, den);
    gat_gemm <<<N_NODES / 16, BT, 0, stream>>>(x, Wb, t);
    gat_alar <<<(HEADS * N_NODES + BT - 1) / BT, BT, 0, stream>>>(t, As, Al, Ar);
    gat_emax <<<(HEADS * E + BT - 1) / BT, BT, 0, stream>>>(src, dst, Al, Ar, m, E);
    gat_eden <<<(HEADS * E + BT - 1) / BT, BT, 0, stream>>>(src, dst, Al, Ar, m, den, E);
    gat_aggr <<<E, BT, 0, stream>>>(src, dst, Al, Ar, m, den, t, out, E);
    gat_elu  <<<(N_NODES * COLS + BT - 1) / BT, BT, 0, stream>>>(out);
}
